// GatedDeltaNet_51075751084058
// MI455X (gfx1250) — compile-verified, matrix operand order checked by reading, not run on hardware
//
#include <hip/hip_runtime.h>
#include <hip/hip_bf16.h>

// ---------------------------------------------------------------------------
// GatedDeltaNet forward for MI455X (gfx1250, wave32, WMMA, async-LDS copies).
// B=2, T=2048, D=1024, H=16, Kd=64, PROJ=1024, KS=4.
//
// bf16 WMMA GEMMs for all projections; chunked delta-rule scan (chunk=32)
// with 64x64 LDS-resident state; tile staging via GLOBAL_LOAD_ASYNC_TO_LDS
// (ASYNCcnt) with double buffering in the big GEMMs.
// ---------------------------------------------------------------------------

typedef __attribute__((ext_vector_type(16))) __bf16 v16bf;
typedef __attribute__((ext_vector_type(8)))  float  v8f;

#define Bsz   2
#define Tsz   2048
#define Dsz   1024
#define Hn    16
#define Kd    64
#define PROJ  1024
#define Mrows (Bsz*Tsz)          // 4096
#define QKVN  (3*PROJ)           // 3072

// ---------------- CDNA5 async global->LDS copy (ASYNCcnt) -------------------
// LDS address comes from the vdst VGPR plus the offset; per-lane 16B copy, no VGPR data.
// LDS byte offset = low 32 bits of the generic pointer (ISA 10.2 aperture map).

__device__ inline void async_ld_b128(void* lds_dst, const void* gsrc) {
  unsigned           loff = (unsigned)(unsigned long long)(uintptr_t)lds_dst;
  unsigned long long gptr = (unsigned long long)(uintptr_t)gsrc;
  asm volatile("global_load_async_to_lds_b128 %0, %1, off"
               :: "v"(loff), "v"(gptr) : "memory");
}
__device__ inline void wait_async0() {
  asm volatile("s_wait_asynccnt 0x0" ::: "memory");
}

// ---------------- WMMA helpers (wave32, 16x16x32 bf16) ----------------------

__device__ inline v8f wmma_bf16(v16bf a, v16bf b, v8f c) {
  return __builtin_amdgcn_wmma_f32_16x16x32_bf16(
      /*neg_a=*/false, a, /*neg_b=*/false, b,
      /*c_mod=*/(short)0, c, /*reuse_a=*/false, /*reuse_b=*/false);
}

// A/B operand fragment load (row = M for A operand, N for pre-transposed B).
// ISA 7.12.2, 16-bit A 16x32:
//   lanes 0-15 : row = lane,    K in [0..7] U [16..23]
//   lanes 16-31: row = lane-16, K in [8..15] U [24..31]
__device__ inline v16bf ldfrag(const __bf16* base, int stride, int lane) {
  const __bf16* p = base + (lane & 15) * stride + ((lane >> 4) << 3);
  union { uint4 u[2]; v16bf v; } r;
  r.u[0] = *(const uint4*)(p);
  r.u[1] = *(const uint4*)(p + 16);
  return r.v;
}

// C/D fragment scatter: VGPR j -> (M = r0 + j (+8 for hi lanes), N = c0+lane&15)
__device__ inline void stcfrag(float* base, int r0, int c0, int stride,
                               int lane, v8f c) {
  int n  = c0 + (lane & 15);
  int rb = r0 + ((lane >> 4) << 3);
  for (int j = 0; j < 8; ++j) base[(rb + j) * stride + n] = c[j];
}

// ---------------- fp32 -> bf16 cast ----------------------------------------

__global__ void cast_f32_bf16_kernel(const float* __restrict__ s,
                                     __bf16* __restrict__ d, int n) {
  int i = blockIdx.x * blockDim.x + threadIdx.x;
  int stride = gridDim.x * blockDim.x;
  for (; i < n; i += stride) d[i] = (__bf16)s[i];
}

// ---------------- big GEMM: C[M,N] = A[M,K](bf16) x Bw[N,K]^T(bf16) ---------
// Block tile 128x256, 8 waves x (64x64 WMMA macro-tile). Double-buffered LDS
// staged entirely with global_load_async_to_lds_b128.

#define TILE_M 128
#define TILE_N 256
#define TILE_K 32
#define LDSA   40   // padded row stride (bf16) -> 80B rows, conflict-free
#define LDSB   40

__global__ __launch_bounds__(256)
void gemm_bf16_kernel(const __bf16* __restrict__ A,
                      const __bf16* __restrict__ Bw,
                      float* __restrict__ C, int N, int K) {
  __shared__ __attribute__((aligned(16))) __bf16 As[2][TILE_M * LDSA];
  __shared__ __attribute__((aligned(16))) __bf16 Bs[2][TILE_N * LDSB];

  const int tid  = threadIdx.x;
  const int lane = tid & 31;
  const int wave = tid >> 5;   // 0..7
  const int wm   = wave & 1;   // 2 waves along M
  const int wn   = wave >> 1;  // 4 waves along N
  const int m0b  = blockIdx.y * TILE_M;
  const int n0b  = blockIdx.x * TILE_N;

  v8f acc[4][4];
  for (int i = 0; i < 4; ++i)
    for (int j = 0; j < 4; ++j) acc[i][j] = v8f{};

  // Issue one K-slice of async copies into buffer `buf`.
  auto issue = [&](int buf, int k0) {
    // A tile: 128x32 bf16 = 512 x 16B chunks, 2 per thread.
    for (int c = tid; c < (TILE_M * TILE_K) / 8; c += 256) {
      int row = c >> 2, seg = c & 3;
      async_ld_b128(&As[buf][row * LDSA + seg * 8],
                    A + (size_t)(m0b + row) * K + k0 + seg * 8);
    }
    // B tile: 256x32 bf16, 4 chunks per thread.
    for (int c = tid; c < (TILE_N * TILE_K) / 8; c += 256) {
      int row = c >> 2, seg = c & 3;
      async_ld_b128(&Bs[buf][row * LDSB + seg * 8],
                    Bw + (size_t)(n0b + row) * K + k0 + seg * 8);
    }
  };

  const int Kiter = K / TILE_K;
  issue(0, 0);
  for (int kb = 0; kb < Kiter; ++kb) {
    wait_async0();        // current buffer's copies have landed in LDS
    __syncthreads();      // visible to all waves; prev buffer readers done
    if (kb + 1 < Kiter) issue((kb + 1) & 1, (kb + 1) * TILE_K);  // overlap

    const __bf16* as = As[kb & 1];
    const __bf16* bs = Bs[kb & 1];
    v16bf af[4];
    for (int mi = 0; mi < 4; ++mi)
      af[mi] = ldfrag(as + (wm * 64 + mi * 16) * LDSA, LDSA, lane);
    for (int ni = 0; ni < 4; ++ni) {
      v16bf bf = ldfrag(bs + (wn * 64 + ni * 16) * LDSB, LDSB, lane);
      for (int mi = 0; mi < 4; ++mi)
        acc[mi][ni] = wmma_bf16(af[mi], bf, acc[mi][ni]);
    }
    __syncthreads();
  }

  for (int mi = 0; mi < 4; ++mi)
    for (int ni = 0; ni < 4; ++ni) {
      int n  = n0b + wn * 64 + ni * 16 + (lane & 15);
      int rb = m0b + wm * 64 + mi * 16 + ((lane >> 4) << 3);
      for (int j = 0; j < 8; ++j)
        C[(size_t)(rb + j) * N + n] = acc[mi][ni][j];
    }
}

// ---------------- g / beta projection (skinny, fp32 VALU) -------------------

__global__ void gb_kernel(const float* __restrict__ x,
                          const float* __restrict__ Wg,
                          const float* __restrict__ Wb,
                          float* __restrict__ g, float* __restrict__ beta) {
  int idx = blockIdx.x * blockDim.x + threadIdx.x;  // 4096 * 32
  if (idx >= Mrows * 2 * Hn) return;
  int token = idx >> 5;
  int r     = idx & 31;
  int head  = r & 15;
  bool isg  = (r < 16);
  const float* w  = (isg ? Wg : Wb) + head * Dsz;
  const float* xr = x + (size_t)token * Dsz;
  float acc = 0.f;
  for (int i = 0; i < Dsz; ++i) acc = fmaf(xr[i], w[i], acc);
  if (isg) {
    float z = acc;  // log_sigmoid, numerically stable
    g[token * Hn + head] = fminf(z, 0.f) - log1pf(__expf(-fabsf(z)));
  } else {
    beta[token * Hn + head] = 1.f / (1.f + __expf(-acc));
  }
}

// ---------------- causal depthwise conv + k-norm + q-scale ------------------
// One block (1 wave) per (b,t,h); lane covers channels d, d+32.

__global__ __launch_bounds__(32)
void conv_kernel(const float* __restrict__ qkv,
                 const float* __restrict__ cwq, const float* __restrict__ cbq,
                 const float* __restrict__ cwk, const float* __restrict__ cbk,
                 const float* __restrict__ cwv, const float* __restrict__ cbv,
                 __bf16* __restrict__ qo, __bf16* __restrict__ ko,
                 __bf16* __restrict__ vo) {
  int blk  = blockIdx.x;            // b*T*H
  int h    = blk & (Hn - 1);
  int bt   = blk >> 4;
  int t    = bt & (Tsz - 1);
  int b    = bt >> 11;
  int lane = threadIdx.x;

  float qc[2], kc[2], vc[2];
  for (int s = 0; s < 2; ++s) {
    int d = lane + s * 32;
    int c = h * Kd + d;
    float aq = cbq[c], ak = cbk[c], av = cbv[c];
    for (int i = 0; i < 4; ++i) {
      int tt = t - 3 + i;
      if (tt < 0) continue;  // causal left zero-pad (per batch)
      const float* row = qkv + (size_t)(b * Tsz + tt) * QKVN;
      aq = fmaf(cwq[c * 4 + i], row[c],            aq);
      ak = fmaf(cwk[c * 4 + i], row[PROJ + c],     ak);
      av = fmaf(cwv[c * 4 + i], row[2 * PROJ + c], av);
    }
    qc[s] = aq; kc[s] = ak; vc[s] = av;
  }
  // ||k|| over the 64 head channels: wave32 butterfly reduction.
  float ss = kc[0] * kc[0] + kc[1] * kc[1];
  for (int off = 16; off; off >>= 1) ss += __shfl_xor(ss, off, 32);
  float rn = 1.f / fmaxf(sqrtf(ss), 1e-12f);

  size_t base = ((size_t)(b * Tsz + t) * Hn + h) * Kd;
  for (int s = 0; s < 2; ++s) {
    int d = lane + s * 32;
    qo[base + d] = (__bf16)(qc[s] * 0.125f);   // Kd^-0.5 = 1/8
    ko[base + d] = (__bf16)(kc[s] * rn);
    vo[base + d] = (__bf16)(vc[s]);
  }
}

// ---------------- chunked gated delta-rule scan -----------------------------
// One wave per (b,h); 64x64 state in LDS (f32 master + bf16 S^T mirror so the
// B operand of every WMMA is a contiguous ds_load_b128 pair). Chunk C=32:
//   Aq = Q~ K^T, Akk = K~ K^T                 (scores; decay applied EW)
//   U  = beta (V - K~ S);  U -= Pstrict U     (one Neumann term of (I+A)^-1)
//   O  = Q~ S + Pincl U
//   S  = e^{g_last} S + K-tilde^T U           (rank-32 WMMA update)

__global__ __launch_bounds__(32)
void delta_scan_kernel(const __bf16* __restrict__ qg_,
                       const __bf16* __restrict__ kg_,
                       const __bf16* __restrict__ vg_,
                       const float* __restrict__ g_,
                       const float* __restrict__ beta_,
                       __bf16* __restrict__ og_) {
  const int bh   = blockIdx.x;  // 0..31
  const int b    = bh >> 4;
  const int h    = bh & 15;
  const int lane = threadIdx.x;

  __shared__ __attribute__((aligned(16))) float  S  [Kd * Kd];   // 16 KB
  __shared__ __attribute__((aligned(16))) __bf16 STb[Kd * Kd];   // S^T, 8 KB
  __shared__ __attribute__((aligned(16))) __bf16 Qg [32 * Kd];
  __shared__ __attribute__((aligned(16))) __bf16 Kg [32 * Kd];
  __shared__ __attribute__((aligned(16))) __bf16 Ks [32 * Kd];
  __shared__ __attribute__((aligned(16))) __bf16 Vs [32 * Kd];
  __shared__ __attribute__((aligned(16))) __bf16 KTt[Kd * 32];   // k-tilde^T
  __shared__ __attribute__((aligned(16))) float  Aqf[32 * 32];
  __shared__ __attribute__((aligned(16))) float  Akf[32 * 32];
  __shared__ __attribute__((aligned(16))) __bf16 Pqi[32 * 32];   // incl. diag
  __shared__ __attribute__((aligned(16))) __bf16 Pks[32 * 32];   // strict
  __shared__ __attribute__((aligned(16))) __bf16 Ub [32 * Kd];
  __shared__ __attribute__((aligned(16))) __bf16 UTb[Kd * 32];
  __shared__ __attribute__((aligned(16))) float  tmpf[32 * Kd];  // 8 KB
  __shared__ float garr[32], gcl[32], bal[32];

  for (int i = lane; i < Kd * Kd; i += 32) { S[i] = 0.f; STb[i] = (__bf16)0.f; }
  __syncthreads();

  for (int t0 = 0; t0 < Tsz; t0 += 32) {
    // ---- load chunk rows straight into LDS (async, no VGPR round-trip) ----
    {
      int t = lane;
      size_t base = ((size_t)(b * Tsz + t0 + t) * Hn + h) * Kd;
      for (int s = 0; s < 8; ++s) {
        async_ld_b128(Qg + t * Kd + s * 8, qg_ + base + s * 8);
        async_ld_b128(Ks + t * Kd + s * 8, kg_ + base + s * 8);
        async_ld_b128(Vs + t * Kd + s * 8, vg_ + base + s * 8);
      }
      garr[t] = g_   [(size_t)(b * Tsz + t0 + t) * Hn + h];
      bal[t]  = beta_[(size_t)(b * Tsz + t0 + t) * Hn + h];
    }
    wait_async0();
    __syncthreads();
    {  // inclusive cumulative log-decay
      float acc = 0.f;
      for (int s = 0; s <= lane; ++s) acc += garr[s];
      gcl[lane] = acc;
    }
    __syncthreads();
    const float glast = gcl[31];
    {  // bake decay into operands: Q~ = q e^{gc}, K~ = k e^{gc},
       // k-tilde^T[d][t] = k beta e^{glast-gc}
      int t = lane;
      float eg  = __expf(gcl[t]);
      float ekt = bal[t] * __expf(glast - gcl[t]);
      for (int d = 0; d < Kd; ++d) {
        float qv = (float)Qg[t * Kd + d];
        float kv = (float)Ks[t * Kd + d];
        Qg[t * Kd + d]  = (__bf16)(qv * eg);
        Kg[t * Kd + d]  = (__bf16)(kv * eg);
        KTt[d * 32 + t] = (__bf16)(kv * ekt);
      }
    }
    __syncthreads();

    // ---- scores Aq = Q~ K^T, Akk = K~ K^T (K-dim 64) ----
    for (int mi = 0; mi < 2; ++mi)
      for (int ni = 0; ni < 2; ++ni) {
        v8f cq = v8f{}, ck = v8f{};
        for (int k2 = 0; k2 < 2; ++k2) {
          v16bf aq = ldfrag(Qg + mi * 16 * Kd + k2 * 32, Kd, lane);
          v16bf ak = ldfrag(Kg + mi * 16 * Kd + k2 * 32, Kd, lane);
          v16bf bb = ldfrag(Ks + ni * 16 * Kd + k2 * 32, Kd, lane);
          cq = wmma_bf16(aq, bb, cq);
          ck = wmma_bf16(ak, bb, ck);
        }
        stcfrag(Aqf, mi * 16, ni * 16, 32, lane, cq);
        stcfrag(Akf, mi * 16, ni * 16, 32, lane, ck);
      }
    __syncthreads();
    {  // causal mask + residual decay e^{-gc_j} (+ beta_i on corrections)
      int i = lane;
      for (int j = 0; j < 32; ++j) {
        float e  = __expf(-gcl[j]);
        float aq = Aqf[i * 32 + j] * e;
        float ak = Akf[i * 32 + j] * e * bal[i];
        Pqi[i * 32 + j] = (j <= i) ? (__bf16)aq : (__bf16)0.f;
        Pks[i * 32 + j] = (j <  i) ? (__bf16)ak : (__bf16)0.f;
      }
    }
    __syncthreads();

    // ---- readout K~ S (via S^T mirror) ----
    for (int mi = 0; mi < 2; ++mi)
      for (int ni = 0; ni < 4; ++ni) {
        v8f c = v8f{};
        for (int k2 = 0; k2 < 2; ++k2) {
          v16bf a  = ldfrag(Kg  + mi * 16 * Kd + k2 * 32, Kd, lane);
          v16bf bb = ldfrag(STb + ni * 16 * Kd + k2 * 32, Kd, lane);
          c = wmma_bf16(a, bb, c);
        }
        stcfrag(tmpf, mi * 16, ni * 16, Kd, lane, c);
      }
    __syncthreads();
    {  // U0 = beta (V - K~ S)
      int t = lane;
      for (int d = 0; d < Kd; ++d) {
        float u = bal[t] * ((float)Vs[t * Kd + d] - tmpf[t * Kd + d]);
        Ub [t * Kd + d] = (__bf16)u;
        UTb[d * 32 + t] = (__bf16)u;
      }
    }
    __syncthreads();
    // ---- first-order correction U -= Pstrict U ----
    for (int mi = 0; mi < 2; ++mi)
      for (int ni = 0; ni < 4; ++ni) {
        v8f c = v8f{};
        v16bf a  = ldfrag(Pks + mi * 16 * 32, 32, lane);
        v16bf bb = ldfrag(UTb + ni * 16 * 32, 32, lane);
        c = wmma_bf16(a, bb, c);
        stcfrag(tmpf, mi * 16, ni * 16, Kd, lane, c);
      }
    __syncthreads();
    {
      int t = lane;
      for (int d = 0; d < Kd; ++d) {
        float u = (float)Ub[t * Kd + d] - tmpf[t * Kd + d];
        Ub [t * Kd + d] = (__bf16)u;
        UTb[d * 32 + t] = (__bf16)u;
      }
    }
    __syncthreads();

    // ---- O = Q~ S + Pincl U  -> global (bf16, feeds output GEMM) ----
    for (int mi = 0; mi < 2; ++mi)
      for (int ni = 0; ni < 4; ++ni) {
        v8f c = v8f{};
        for (int k2 = 0; k2 < 2; ++k2) {
          v16bf a  = ldfrag(Qg  + mi * 16 * Kd + k2 * 32, Kd, lane);
          v16bf bb = ldfrag(STb + ni * 16 * Kd + k2 * 32, Kd, lane);
          c = wmma_bf16(a, bb, c);
        }
        {
          v16bf a  = ldfrag(Pqi + mi * 16 * 32, 32, lane);
          v16bf bb = ldfrag(UTb + ni * 16 * 32, 32, lane);
          c = wmma_bf16(a, bb, c);
        }
        int n  = ni * 16 + (lane & 15);
        int rb = mi * 16 + ((lane >> 4) << 3);
        for (int j = 0; j < 8; ++j) {
          int t = rb + j;
          og_[(size_t)(b * Tsz + t0 + t) * PROJ + h * Kd + n] = (__bf16)c[j];
        }
      }
    __syncthreads();

    // ---- state update S = e^{glast} S + k-tilde^T U (rank-32) ----
    const float edec = __expf(glast);
    for (int mi = 0; mi < 4; ++mi)
      for (int ni = 0; ni < 4; ++ni) {
        v8f c;
        int n  = ni * 16 + (lane & 15);
        int rb = mi * 16 + ((lane >> 4) << 3);
        for (int j = 0; j < 8; ++j) c[j] = S[(rb + j) * Kd + n] * edec;
        v16bf a  = ldfrag(KTt + mi * 16 * 32, 32, lane);
        v16bf bb = ldfrag(UTb + ni * 16 * 32, 32, lane);
        c = wmma_bf16(a, bb, c);
        for (int j = 0; j < 8; ++j) {
          S  [(rb + j) * Kd + n] = c[j];
          STb[n * Kd + (rb + j)] = (__bf16)c[j];
        }
      }
    __syncthreads();
  }
}

// ---------------------------------------------------------------------------

extern "C" void kernel_launch(void* const* d_in, const int* in_sizes, int n_in,
                              void* d_out, int out_size, void* d_ws,
                              size_t ws_size, hipStream_t stream) {
  (void)in_sizes; (void)n_in; (void)out_size; (void)ws_size;

  const float* x   = (const float*)d_in[0];
  const float* Wq  = (const float*)d_in[1];
  const float* Wk  = (const float*)d_in[2];
  const float* Wv  = (const float*)d_in[3];
  const float* Wo  = (const float*)d_in[4];
  const float* Wg  = (const float*)d_in[5];
  const float* Wb  = (const float*)d_in[6];
  const float* cwq = (const float*)d_in[7];
  const float* cbq = (const float*)d_in[8];
  const float* cwk = (const float*)d_in[9];
  const float* cbk = (const float*)d_in[10];
  const float* cwv = (const float*)d_in[11];
  const float* cbv = (const float*)d_in[12];
  float* out = (float*)d_out;

  // Workspace layout (bytes).
  char* ws = (char*)d_ws;
  size_t off = 0;
  __bf16* x_bf   = (__bf16*)(ws + off); off += (size_t)Mrows * Dsz * 2;   // 8 MB
  __bf16* wqkv   = (__bf16*)(ws + off); off += (size_t)QKVN * Dsz * 2;    // 6 MB
  __bf16* wo_bf  = (__bf16*)(ws + off); off += (size_t)Dsz * PROJ * 2;    // 2 MB
  float*  qkv_f  = (float*) (ws + off); off += (size_t)Mrows * QKVN * 4;  // 48 MB
  float*  g_f    = (float*) (ws + off); off += (size_t)Mrows * Hn * 4;
  float*  be_f   = (float*) (ws + off); off += (size_t)Mrows * Hn * 4;
  __bf16* q_bf   = (__bf16*)(ws + off); off += (size_t)Mrows * PROJ * 2;  // 8 MB
  __bf16* k_bf   = (__bf16*)(ws + off); off += (size_t)Mrows * PROJ * 2;
  __bf16* v_bf   = (__bf16*)(ws + off); off += (size_t)Mrows * PROJ * 2;
  __bf16* o_bf   = (__bf16*)(ws + off); off += (size_t)Mrows * PROJ * 2;

  // 1) fp32 -> bf16 casts (x, [Wq;Wk;Wv] concatenated, Wo).
  const int WN = PROJ * Dsz;
  cast_f32_bf16_kernel<<<2048, 256, 0, stream>>>(x, x_bf, Mrows * Dsz);
  cast_f32_bf16_kernel<<<1024, 256, 0, stream>>>(Wq, wqkv,          WN);
  cast_f32_bf16_kernel<<<1024, 256, 0, stream>>>(Wk, wqkv + WN,     WN);
  cast_f32_bf16_kernel<<<1024, 256, 0, stream>>>(Wv, wqkv + 2 * WN, WN);
  cast_f32_bf16_kernel<<<1024, 256, 0, stream>>>(Wo, wo_bf,         WN);

  // 2) fused QKV projection: qkv[4096,3072] = x @ [Wq;Wk;Wv]^T  (WMMA bf16)
  gemm_bf16_kernel<<<dim3(QKVN / TILE_N, Mrows / TILE_M), 256, 0, stream>>>(
      x_bf, wqkv, qkv_f, QKVN, Dsz);

  // 3) g = log_sigmoid(x Wg^T), beta = sigmoid(x Wb^T)
  gb_kernel<<<(Mrows * 32 + 255) / 256, 256, 0, stream>>>(x, Wg, Wb, g_f, be_f);

  // 4) causal depthwise conv + k-norm + q-scale -> bf16 [B,T,H,Kd]
  conv_kernel<<<Bsz * Tsz * Hn, 32, 0, stream>>>(
      qkv_f, cwq, cbq, cwk, cbk, cwv, cbv, q_bf, k_bf, v_bf);

  // 5) chunked gated delta-rule scan (32 waves, WMMA-resident state)
  delta_scan_kernel<<<Bsz * Hn, 32, 0, stream>>>(q_bf, k_bf, v_bf, g_f, be_f,
                                                 o_bf);

  // 6) output projection: out = o @ Wo^T  (WMMA bf16)
  gemm_bf16_kernel<<<dim3(Dsz / TILE_N, Mrows / TILE_M), 256, 0, stream>>>(
      o_bf, wo_bf, out, Dsz, PROJ);
}
